// MultiHeadAttention_25348896981340
// MI455X (gfx1250) — matrix operand order repaired and checked by reading, not compiled, not run on hardware
//
#include <hip/hip_runtime.h>
#include <hip/hip_bf16.h>
#include <stdint.h>

#define B_ 4
#define L_ 2048
#define D_ 1024

typedef __attribute__((ext_vector_type(16))) __bf16 v16bf;
typedef __attribute__((ext_vector_type(8)))  float  v8f;

// Explicit global-address-space pointers so fragment loads lower to
// global_load_b128 (LOADcnt only) instead of flat_load_b128 (LOAD+DS cnt).
typedef const uint16_t __attribute__((address_space(1)))* gu16p;
typedef const uint32_t __attribute__((address_space(1)))* gu32p;

union FragBF { v16bf v; uint32_t u[8]; };
union AccF   { v8f   v; float    f[8]; };

// round-to-nearest-even f32 -> bf16
__device__ __forceinline__ uint16_t f2bf(float x) {
    union { float f; uint32_t u; } a; a.f = x;
    return (uint16_t)((a.u + 0x7FFFu + ((a.u >> 16) & 1u)) >> 16);
}
__device__ __forceinline__ uint32_t pack_bf16(float lo, float hi) {
    return ((uint32_t)f2bf(hi) << 16) | (uint32_t)f2bf(lo);
}

// -------------------------------------------------------------------------
// Elementwise f32 -> bf16 (bandwidth-bound, 4 elems/thread)
// -------------------------------------------------------------------------
__global__ __launch_bounds__(256) void cvt_bf16_k(const float* __restrict__ in,
                                                  uint16_t* __restrict__ out) {
    const size_t i = ((size_t)blockIdx.x * blockDim.x + threadIdx.x) * 4;
    const float4 v = *reinterpret_cast<const float4*>(in + i);
    uint2 r;
    r.x = pack_bf16(v.x, v.y);
    r.y = pack_bf16(v.z, v.w);
    *reinterpret_cast<uint2*>(out + i) = r;
}

// -------------------------------------------------------------------------
// Transpose-convert: W[k][n] f32 (1024x1024) -> Wt[n][k] bf16. LDS-tiled.
// -------------------------------------------------------------------------
__global__ __launch_bounds__(256) void wtrans_k(const float* __restrict__ W,
                                                uint16_t* __restrict__ Wt) {
    __shared__ uint16_t tile[32][33];
    const int bx = blockIdx.x & 31;         // n-tile
    const int by = blockIdx.x >> 5;         // k-tile
    const int tx = threadIdx.x & 31;
    const int ty = threadIdx.x >> 5;        // 0..7
#pragma unroll
    for (int j = 0; j < 4; ++j) {
        const int k = by * 32 + ty + j * 8;
        tile[ty + j * 8][tx] = f2bf(W[(size_t)k * D_ + bx * 32 + tx]);
    }
    __syncthreads();
#pragma unroll
    for (int j = 0; j < 4; ++j) {
        const int n = bx * 32 + ty + j * 8;
        Wt[(size_t)n * D_ + by * 32 + tx] = tile[tx][ty + j * 8];
    }
}

// -------------------------------------------------------------------------
// Unified register-blocked bf16 WMMA GEMM: C = A(MxK) @ Bt(NxK)^T
// Wave tile 64x64: 4x4 = 16 WMMA accumulators, 4x reuse of A and B frags,
// 16 b128 global loads per 16 WMMAs. Dedicated AS(1) running pointers
// (half-offset folded in) -> pointer bumps never touch WMMA operand regs.
// EPI: 0 bias->bf16 | 1 bias->f32 | 2 scale+mask->f32 (batched)
//      3 ->bf16 (batched) | 4 bias->bf16 transposed per-batch (for V^T)
// -------------------------------------------------------------------------
template <int EPI>
__global__ __launch_bounds__(256) void wgemm_k(
    const uint16_t* __restrict__ A, int lda, size_t batchA,
    const uint16_t* __restrict__ Bt, size_t batchB,
    const float* __restrict__ bias, const int* __restrict__ mask,
    float* __restrict__ Cf, uint16_t* __restrict__ Cbf, size_t batchC,
    int M, int N, int K) {
    const int lane = threadIdx.x & 31;
    const int wave = blockIdx.x * 8 + (threadIdx.x >> 5);
    const int ntn  = N >> 6;                 // 64-wide N tiles
    const int tileM = wave / ntn;
    const int tileN = wave % ntn;
    if (tileM * 64 >= M) return;
    const int half = lane >> 4;
    const int l16  = lane & 15;
    const int b    = blockIdx.y;

    const uint16_t* Ab = A  + (size_t)b * batchA;
    const uint16_t* Bb = Bt + (size_t)b * batchB;
    const int m0 = tileM * 64 + l16;
    const int n0 = tileN * 64 + l16;

    // Running global pointers: A lane offset (half*8) and B lane offset
    // (half*8) folded in once; K progression = +32 elems (64 B) per iter.
    gu16p pa[4];
    gu16p pb[4];
#pragma unroll
    for (int t = 0; t < 4; ++t) {
        pa[t] = (gu16p)(uintptr_t)(Ab + (size_t)(m0 + t * 16) * lda + (half << 3));
        pb[t] = (gu16p)(uintptr_t)(Bb + (size_t)(n0 + t * 16) * K   + (half << 3));
    }

    AccF acc[4][4];
#pragma unroll
    for (int mt = 0; mt < 4; ++mt)
#pragma unroll
        for (int nt = 0; nt < 4; ++nt)
#pragma unroll
            for (int i = 0; i < 8; ++i) acc[mt][nt].f[i] = 0.f;

    for (int kb = 0; kb < K; kb += 32) {
        FragBF a[4], bf[4];
#pragma unroll
        for (int t = 0; t < 4; ++t) {
#pragma unroll
            for (int v = 0; v < 8; ++v) {
                // A: ushort offsets {0,2,4,6, 16,18,20,22} -> two b128 loads
                const int oa = ((v >> 2) << 4) + ((v & 3) << 1);
                a[t].u[v] = *(gu32p)(pa[t] + oa);
                // B: ushort offsets {0,2,4,6, 16,18,20,22} -> two b128 loads
                bf[t].u[v] = *(gu32p)(pb[t] + oa);
            }
        }
#pragma unroll
        for (int mt = 0; mt < 4; ++mt)
#pragma unroll
            for (int nt = 0; nt < 4; ++nt)
                acc[mt][nt].v = __builtin_amdgcn_wmma_f32_16x16x32_bf16(
                    false, a[mt].v, false, bf[nt].v, (short)0, acc[mt][nt].v,
                    false, false);
#pragma unroll
        for (int t = 0; t < 4; ++t) { pa[t] += 32; pb[t] += 32; }
    }

#pragma unroll
    for (int mt = 0; mt < 4; ++mt)
#pragma unroll
        for (int nt = 0; nt < 4; ++nt)
#pragma unroll
            for (int v = 0; v < 8; ++v) {
                const int row = tileM * 64 + mt * 16 + (half << 3) + v;
                const int col = tileN * 64 + nt * 16 + l16;
                float val = acc[mt][nt].f[v];
                if (EPI == 0) {
                    Cbf[(size_t)row * N + col] = f2bf(val + bias[col]);
                } else if (EPI == 1) {
                    Cf[(size_t)row * N + col] = val + bias[col];
                } else if (EPI == 2) {
                    const size_t idx = (size_t)b * batchC + (size_t)row * N + col;
                    val *= 0.03125f;                       // 1/sqrt(1024)
                    Cf[idx] = (mask[idx] == 0) ? -1.0e20f : val;
                } else if (EPI == 3) {
                    Cbf[(size_t)b * batchC + (size_t)row * N + col] = f2bf(val);
                } else {                                   // EPI 4: V^T store
                    const int vb = row >> 11, pos = row & (L_ - 1);
                    Cbf[((size_t)vb * N + col) * L_ + pos] = f2bf(val + bias[col]);
                }
            }
}

// -------------------------------------------------------------------------
// Row softmax over Lk=2048; one wave per row; bf16 P written into the
// second half of the same f32 row (full row is read before any write).
// -------------------------------------------------------------------------
__global__ __launch_bounds__(256) void softmax_k(float* __restrict__ S) {
    const int lane = threadIdx.x & 31;
    const int row  = blockIdx.x * 8 + (threadIdx.x >> 5);
    float* Srow = S + (size_t)row * L_;

    float vals[64];
    float mx = -3.0e38f;
#pragma unroll
    for (int j = 0; j < 64; ++j) {
        vals[j] = Srow[j * 32 + lane];
        mx = fmaxf(mx, vals[j]);
    }
#pragma unroll
    for (int off = 16; off; off >>= 1) mx = fmaxf(mx, __shfl_xor(mx, off, 32));

    float sum = 0.f;
#pragma unroll
    for (int j = 0; j < 64; ++j) {
        vals[j] = __expf(vals[j] - mx);
        sum += vals[j];
    }
#pragma unroll
    for (int off = 16; off; off >>= 1) sum += __shfl_xor(sum, off, 32);

    const float inv = 1.0f / sum;
    uint16_t* Prow = reinterpret_cast<uint16_t*>(Srow) + L_;
#pragma unroll
    for (int j = 0; j < 64; ++j) Prow[j * 32 + lane] = f2bf(vals[j] * inv);
}

// -------------------------------------------------------------------------
extern "C" void kernel_launch(void* const* d_in, const int* in_sizes, int n_in,
                              void* d_out, int out_size, void* d_ws, size_t ws_size,
                              hipStream_t stream) {
    const float* query = (const float*)d_in[0];
    const float* key   = (const float*)d_in[1];
    const float* value = (const float*)d_in[2];
    const int*   mask  = (const int*)  d_in[3];
    const float* Wq = (const float*)d_in[4];  const float* bq = (const float*)d_in[5];
    const float* Wk = (const float*)d_in[6];  const float* bk = (const float*)d_in[7];
    const float* Wv = (const float*)d_in[8];  const float* bv = (const float*)d_in[9];
    const float* Wo = (const float*)d_in[10]; const float* bo = (const float*)d_in[11];
    float* out = (float*)d_out;

    const size_t MB = 1024ull * 1024ull;
    uint8_t* ws = (uint8_t*)d_ws;
    // S (64 MB) overlaps the bf16 inputs (dead after the projection GEMMs).
    float*    S   = (float*)   (ws);               // [0,64) MB
    uint16_t* Xq  = (uint16_t*)(ws);               // [0,16) MB  (dead before S written)
    uint16_t* Xk  = (uint16_t*)(ws + 16 * MB);     // [16,32) MB
    uint16_t* Xv  = (uint16_t*)(ws + 32 * MB);     // [32,48) MB
    uint16_t* Qbf = (uint16_t*)(ws + 64 * MB);     // [64,80) MB
    uint16_t* Kbf = (uint16_t*)(ws + 80 * MB);     // [80,96) MB
    uint16_t* Vt  = (uint16_t*)(ws + 96 * MB);     // [96,112) MB  V^T per batch (d,pos)
    uint16_t* Wtq = (uint16_t*)(ws + 112 * MB);    // 2 MB each
    uint16_t* Wtk = (uint16_t*)(ws + 114 * MB);
    uint16_t* Wtv = (uint16_t*)(ws + 116 * MB);
    uint16_t* Wto = (uint16_t*)(ws + 118 * MB);
    uint16_t* AO  = (uint16_t*)(ws + 120 * MB);    // [120,136) MB

    const int M = B_ * L_;                          // 8192
    dim3 blk(256);
    const int cvt_blocks = (M * D_ / 4) / 256;      // 8192
    const size_t LD = (size_t)L_ * D_;
    const size_t LL = (size_t)L_ * L_;

    cvt_bf16_k<<<cvt_blocks, blk, 0, stream>>>(query, Xq);
    cvt_bf16_k<<<cvt_blocks, blk, 0, stream>>>(key,   Xk);
    cvt_bf16_k<<<cvt_blocks, blk, 0, stream>>>(value, Xv);
    wtrans_k<<<1024, blk, 0, stream>>>(Wq, Wtq);
    wtrans_k<<<1024, blk, 0, stream>>>(Wk, Wtk);
    wtrans_k<<<1024, blk, 0, stream>>>(Wv, Wtv);
    wtrans_k<<<1024, blk, 0, stream>>>(Wo, Wto);

    // Projections: M=8192, N=K=1024 -> (128*16)/8 = 256 blocks
    wgemm_k<0><<<dim3(256, 1), blk, 0, stream>>>(Xq, D_, 0, Wtq, 0, bq, nullptr,
                                                 nullptr, Qbf, 0, M, D_, D_);
    wgemm_k<0><<<dim3(256, 1), blk, 0, stream>>>(Xk, D_, 0, Wtk, 0, bk, nullptr,
                                                 nullptr, Kbf, 0, M, D_, D_);
    wgemm_k<4><<<dim3(256, 1), blk, 0, stream>>>(Xv, D_, 0, Wtv, 0, bv, nullptr,
                                                 nullptr, Vt, 0, M, D_, D_);
    // Scores: per batch M=N=2048, K=1024 -> (32*32)/8 = 128 blocks, grid.y=4
    wgemm_k<2><<<dim3(128, B_), blk, 0, stream>>>(Qbf, D_, LD, Kbf, LD, nullptr,
                                                  mask, S, nullptr, LL, L_, L_, D_);
    softmax_k<<<(B_ * L_) / 8, blk, 0, stream>>>(S);
    // P @ V: per batch M=2048, N=1024, K=2048 -> (32*16)/8 = 64 blocks, grid.y=4
    wgemm_k<3><<<dim3(64, B_), blk, 0, stream>>>(
        (const uint16_t*)S + L_, 2 * L_, (size_t)L_ * 2 * L_,
        Vt, (size_t)D_ * L_, nullptr, nullptr, nullptr, AO, LD, L_, D_, L_);
    // Final projection: f32 out
    wgemm_k<1><<<dim3(256, 1), blk, 0, stream>>>(AO, D_, 0, Wto, 0, bo, nullptr,
                                                 out, nullptr, 0, M, D_, D_);
}
